// CausalSelfAttention_14980845928591
// MI455X (gfx1250) — matrix operand order repaired and checked by reading, not compiled, not run on hardware
//
#include <hip/hip_runtime.h>
#include <math.h>

// ---------------- problem constants ----------------
#define NH  16
#define HD  64
#define CC  1024          // C = NH*HD
#define TT  2048
#define BB  4
#define BT  (BB*TT)       // 8192 rows

typedef __bf16 bf16;
typedef __attribute__((ext_vector_type(16))) __bf16 v16bf;
typedef __attribute__((ext_vector_type(8)))  __bf16 v8bf;
typedef __attribute__((ext_vector_type(8)))  float  v8f;

union AFrag { v16bf v; v8bf h[2]; };

__device__ __forceinline__ v8f wmma_bf16(v16bf a, v16bf b, v8f c) {
  // D = A(16x32 bf16) * B(32x16 bf16) + C(16x16 f32)
  return __builtin_amdgcn_wmma_f32_16x16x32_bf16(false, a, false, b, (short)0, c, false, false);
}

// ---------------- pack kernels ----------------
__global__ void k_cvt_bf16(const float* __restrict__ src, bf16* __restrict__ dst, long n) {
  long i = (long)blockIdx.x * blockDim.x + threadIdx.x;
  if (i < n) dst[i] = (bf16)src[i];
}

// Pack weight w[K][N] (row-major f32) into per-wave B-fragment order:
// layout [nb][kb][512]; within a 32x16 (KxN) block, element index = lane*16+e,
// lane's column n = nb*16 + (lane&15), k = kb*32 + (lane>>4)*16 + e.
__global__ void k_pack_weight(const float* __restrict__ w, bf16* __restrict__ wp, int K, int N) {
  long i = (long)blockIdx.x * blockDim.x + threadIdx.x;
  long total = (long)K * N;
  if (i >= total) return;
  int  e       = (int)(i & 511);
  long blk     = i >> 9;
  int  kBlocks = K >> 5;
  int  kb   = (int)(blk % kBlocks);
  int  nb   = (int)(blk / kBlocks);
  int  lane = e >> 4;
  int  el   = e & 15;
  int  n = nb * 16 + (lane & 15);
  int  k = kb * 32 + ((lane >> 4) << 4) + el;
  wp[i] = (bf16)w[(size_t)k * N + n];
}

// ---------------- GEMM core: one wave -> 32 rows x 64 cols ----------------
// B fragments reused across both 16-row A tiles: 8 WMMA per 8 fragment loads.
__device__ __forceinline__ void gemm32x64(const bf16* __restrict__ A, int lda,
                                          const bf16* __restrict__ Bp, int kBlocks,
                                          int row0, int nb0, int lane, v8f acc[2][4]) {
  const int m    = lane & 15;
  const int half = lane >> 4;
  const bf16* arow0 = A + (size_t)(row0 + m) * lda;
  const bf16* arow1 = A + (size_t)(row0 + 16 + m) * lda;
  for (int kb = 0; kb < kBlocks; ++kb) {
    AFrag a0, a1;
    a0.h[0] = *(const v8bf*)(arow0 + kb * 32 + half * 16);
    a0.h[1] = *(const v8bf*)(arow0 + kb * 32 + half * 16 + 8);
    a1.h[0] = *(const v8bf*)(arow1 + kb * 32 + half * 16);
    a1.h[1] = *(const v8bf*)(arow1 + kb * 32 + half * 16 + 8);
#pragma unroll
    for (int j = 0; j < 4; ++j) {
      const v16bf bfrag = *(const v16bf*)(Bp + ((size_t)(nb0 + j) * kBlocks + kb) * 512 + lane * 16);
      acc[0][j] = wmma_bf16(a0.v, bfrag, acc[0][j]);
      acc[1][j] = wmma_bf16(a1.v, bfrag, acc[1][j]);
    }
  }
}

// ---------------- QKV projection ----------------
// out col c: [0,C) -> Q (scaled 1/8) [B,H,T,D]; [C,2C) -> K [B,H,T,D]; [2C,3C) -> V^T [B,H,D,T]
__global__ void k_qkv(const bf16* __restrict__ xb, const bf16* __restrict__ wp,
                      bf16* __restrict__ Qb, bf16* __restrict__ Kb, bf16* __restrict__ Vt) {
  const int lane = threadIdx.x & 31;
  const int row0 = blockIdx.x * 32;
  const int nb0  = blockIdx.y * 4;
  v8f acc[2][4] = {};
  gemm32x64(xb, CC, wp, CC / 32, row0, nb0, lane, acc);
  const int m = lane & 15, half = lane >> 4;
#pragma unroll
  for (int mt = 0; mt < 2; ++mt) {
#pragma unroll
    for (int j = 0; j < 4; ++j) {
      int c = (nb0 + j) * 16 + m;
#pragma unroll
      for (int r = 0; r < 8; ++r) {
        int row = row0 + mt * 16 + half * 8 + r;   // C-layout: VGPR r -> row r (+8 lanes 16-31)
        int b = row / TT;
        int t = row - b * TT;
        float v = acc[mt][j][r];
        if (c < CC) {
          int h = c >> 6, d = c & 63;
          Qb[(((size_t)(b * NH + h)) * TT + t) * HD + d] = (bf16)(v * 0.125f); // 1/sqrt(64)
        } else if (c < 2 * CC) {
          int cc2 = c - CC; int h = cc2 >> 6, d = cc2 & 63;
          Kb[(((size_t)(b * NH + h)) * TT + t) * HD + d] = (bf16)v;
        } else {
          int cc2 = c - 2 * CC; int h = cc2 >> 6, d = cc2 & 63;
          Vt[(((size_t)(b * NH + h)) * HD + d) * TT + t] = (bf16)v;            // transposed
        }
      }
    }
  }
}

// ---------------- gate = sigmoid(x @ w_gate + b) ----------------
__global__ void k_gate(const bf16* __restrict__ xb, const bf16* __restrict__ wp,
                       const float* __restrict__ bias, float* __restrict__ gate) {
  const int lane = threadIdx.x & 31;
  const int row0 = blockIdx.x * 32;
  const int nb0  = blockIdx.y * 4;
  v8f acc[2][4] = {};
  gemm32x64(xb, CC, wp, CC / 32, row0, nb0, lane, acc);
  const int m = lane & 15, half = lane >> 4;
#pragma unroll
  for (int mt = 0; mt < 2; ++mt) {
#pragma unroll
    for (int j = 0; j < 4; ++j) {
      int c = (nb0 + j) * 16 + m;
#pragma unroll
      for (int r = 0; r < 8; ++r) {
        int row = row0 + mt * 16 + half * 8 + r;
        float v = acc[mt][j][r] + bias[c];
        gate[(size_t)row * CC + c] = 1.0f / (1.0f + __expf(-v));
      }
    }
  }
}

// ---------------- flash-style causal attention (one wave per 16 q-rows of one head) ----------------
__global__ void k_attn(const bf16* __restrict__ Qb, const bf16* __restrict__ Kb,
                       const bf16* __restrict__ Vt, const float* __restrict__ gate,
                       bf16* __restrict__ ybg) {
  const int lane  = threadIdx.x & 31;
  const int qBase = blockIdx.x * 16;
  const int bh    = blockIdx.y;
  const int b = bh / NH, h = bh - b * NH;

  const bf16* Qh = Qb + (size_t)bh * TT * HD;   // [T][D]
  const bf16* Kh = Kb + (size_t)bh * TT * HD;   // [T][D]
  const bf16* Vh = Vt + (size_t)bh * HD * TT;   // [D][T]

  const int m = lane & 15, half = lane >> 4;
  const int rbase = half * 8;

  // Q A-fragments for d-chunks {0..31}, {32..63}
  AFrag qa[2];
  const bf16* qrow = Qh + (size_t)(qBase + m) * HD;
#pragma unroll
  for (int kb = 0; kb < 2; ++kb) {
    qa[kb].h[0] = *(const v8bf*)(qrow + kb * 32 + half * 16);
    qa[kb].h[1] = *(const v8bf*)(qrow + kb * 32 + half * 16 + 8);
  }

  v8f yacc[4] = {};                 // y tile: 16 x 64 (4 n-tiles of 16)
  float mrow[8], lrow[8];
#pragma unroll
  for (int r = 0; r < 8; ++r) { mrow[r] = -1e30f; lrow[r] = 0.0f; }

  __shared__ __align__(32) bf16 Pld[16 * 32];   // P staging (C-layout -> A-fragment reshape)

  for (int sBase = 0; sBase <= qBase + 15; sBase += 32) {
    // S = Q*K^T for keys [sBase, sBase+32)
    v8f s0 = {}, s1 = {};
#pragma unroll
    for (int kb = 0; kb < 2; ++kb) {
      const v16bf kf0 = *(const v16bf*)(Kh + (size_t)(sBase + m)      * HD + kb * 32 + half * 16);
      const v16bf kf1 = *(const v16bf*)(Kh + (size_t)(sBase + 16 + m) * HD + kb * 32 + half * 16);
      s0 = wmma_bf16(qa[kb].v, kf0, s0);
      s1 = wmma_bf16(qa[kb].v, kf1, s1);
    }
    // online softmax, per row r (row = qBase + rbase + r, this lane's col = m / m+16)
#pragma unroll
    for (int r = 0; r < 8; ++r) {
      int qrg = qBase + rbase + r;
      float a = s0[r];  if (sBase + m      > qrg) a = -1e30f;   // causal mask
      float c = s1[r];  if (sBase + 16 + m > qrg) c = -1e30f;
      float mx = fmaxf(a, c);
#pragma unroll
      for (int off = 1; off < 16; off <<= 1) mx = fmaxf(mx, __shfl_xor(mx, off, 32));
      float mnew  = fmaxf(mrow[r], mx);
      float alpha = __expf(mrow[r] - mnew);
      mrow[r] = mnew;
      float p0 = __expf(a - mnew);
      float p1 = __expf(c - mnew);
      float rs = p0 + p1;
#pragma unroll
      for (int off = 1; off < 16; off <<= 1) rs += __shfl_xor(rs, off, 32);
      lrow[r] = lrow[r] * alpha + rs;
#pragma unroll
      for (int j = 0; j < 4; ++j) yacc[j][r] *= alpha;
      Pld[(rbase + r) * 32 + m]      = (bf16)p0;
      Pld[(rbase + r) * 32 + 16 + m] = (bf16)p1;
    }
    __syncthreads();   // single-wave WG: LDS per-wave in-order; keeps waits explicit

    // P as A-fragment (16x32), V^T rows as B-fragments: y += P * V
    AFrag pa;
    const bf16* prow = &Pld[m * 32];
    pa.h[0] = *(const v8bf*)(prow + half * 16);
    pa.h[1] = *(const v8bf*)(prow + half * 16 + 8);
#pragma unroll
    for (int j = 0; j < 4; ++j) {
      const v16bf vf = *(const v16bf*)(Vh + (size_t)(j * 16 + m) * TT + sBase + half * 16);
      yacc[j] = wmma_bf16(pa.v, vf, yacc[j]);
    }
    __syncthreads();
  }

  // epilogue: y/l, apply gate, store bf16 for proj GEMM
#pragma unroll
  for (int r = 0; r < 8; ++r) {
    int t = qBase + rbase + r;
    size_t rowflat = (size_t)(b * TT + t);
    float inv = 1.0f / lrow[r];
#pragma unroll
    for (int j = 0; j < 4; ++j) {
      int col = h * HD + j * 16 + m;
      float g = gate[rowflat * CC + col];
      ybg[rowflat * CC + col] = (bf16)(yacc[j][r] * inv * g);
    }
  }
}

// ---------------- output projection ----------------
__global__ void k_proj(const bf16* __restrict__ yb, const bf16* __restrict__ wp,
                       float* __restrict__ out) {
  const int lane = threadIdx.x & 31;
  const int row0 = blockIdx.x * 32;
  const int nb0  = blockIdx.y * 4;
  v8f acc[2][4] = {};
  gemm32x64(yb, CC, wp, CC / 32, row0, nb0, lane, acc);
  const int m = lane & 15, half = lane >> 4;
#pragma unroll
  for (int mt = 0; mt < 2; ++mt) {
#pragma unroll
    for (int j = 0; j < 4; ++j) {
      int c = (nb0 + j) * 16 + m;
#pragma unroll
      for (int r = 0; r < 8; ++r) {
        int row = row0 + mt * 16 + half * 8 + r;
        out[(size_t)row * CC + c] = acc[mt][j][r];
      }
    }
  }
}

// ---------------- launch ----------------
extern "C" void kernel_launch(void* const* d_in, const int* in_sizes, int n_in,
                              void* d_out, int out_size, void* d_ws, size_t ws_size,
                              hipStream_t stream) {
  (void)in_sizes; (void)n_in; (void)out_size; (void)ws_size;
  const float* x      = (const float*)d_in[0];
  const float* w_attn = (const float*)d_in[1];
  const float* w_proj = (const float*)d_in[2];
  const float* w_gate = (const float*)d_in[3];
  const float* b_gate = (const float*)d_in[4];

  char* ws = (char*)d_ws;
  bf16*  xb     = (bf16*)ws;  ws += (size_t)BT * CC * 2;        // 16 MB
  bf16*  wqkvp  = (bf16*)ws;  ws += (size_t)CC * 3 * CC * 2;    //  6 MB
  bf16*  wgatep = (bf16*)ws;  ws += (size_t)CC * CC * 2;        //  2 MB
  bf16*  wprojp = (bf16*)ws;  ws += (size_t)CC * CC * 2;        //  2 MB
  bf16*  Qb     = (bf16*)ws;  ws += (size_t)BT * CC * 2;        // 16 MB
  bf16*  Kb     = (bf16*)ws;  ws += (size_t)BT * CC * 2;        // 16 MB
  bf16*  Vt     = (bf16*)ws;  ws += (size_t)BT * CC * 2;        // 16 MB
  float* gateb  = (float*)ws; ws += (size_t)BT * CC * 4;        // 32 MB
  bf16*  ybg    = (bf16*)ws;  ws += (size_t)BT * CC * 2;        // 16 MB

  long nx = (long)BT * CC;
  k_cvt_bf16<<<dim3((unsigned)((nx + 255) / 256)), 256, 0, stream>>>(x, xb, nx);
  long nwa = (long)CC * 3 * CC;
  k_pack_weight<<<dim3((unsigned)((nwa + 255) / 256)), 256, 0, stream>>>(w_attn, wqkvp, CC, 3 * CC);
  long nw = (long)CC * CC;
  k_pack_weight<<<dim3((unsigned)((nw + 255) / 256)), 256, 0, stream>>>(w_gate, wgatep, CC, CC);
  k_pack_weight<<<dim3((unsigned)((nw + 255) / 256)), 256, 0, stream>>>(w_proj, wprojp, CC, CC);

  k_qkv <<<dim3(BT / 32, (3 * CC) / 64), 32, 0, stream>>>(xb, wqkvp, Qb, Kb, Vt);
  k_gate<<<dim3(BT / 32, CC / 64),       32, 0, stream>>>(xb, wgatep, b_gate, gateb);
  k_attn<<<dim3(TT / 16, BB * NH),       32, 0, stream>>>(Qb, Kb, Vt, gateb, ybg);
  k_proj<<<dim3(BT / 32, CC / 64),       32, 0, stream>>>(ybg, wprojp, (float*)d_out);
}
